// CholeskyMMNet_9311489097934
// MI455X (gfx1250) — matrix operand order repaired and checked by reading, not compiled, not run on hardware
//
#include <hip/hip_runtime.h>
#include <hip/hip_bf16.h>

// ---------------------------------------------------------------------------
// CholeskyMMNet fused kernel for gfx1250 (MI455X), wave32 + WMMA bf16.
//   x = relu(q@W_in+b);  x = relu(x@W_h1+b)+x;  x = relu(x@W_h2+b)+x;
//   e = x@W_out+b;  Lp = e@W_o+b;  L = tril(Lp)+diag(+2);  M = L@L^T
// One block = 128 batch rows, 8 waves, each wave owns 16 rows.
// Activations live in LDS; weights streamed (bf16, pre-transposed) from L2
// via GLOBAL_LOAD_ASYNC_TO_LDS (ASYNCcnt) when available; biases staged in
// LDS so the weight prefetch never serializes on LOADcnt.
// ---------------------------------------------------------------------------

typedef __attribute__((ext_vector_type(16))) __bf16       v16bf;
typedef __attribute__((ext_vector_type(8)))  float        v8f;
typedef __attribute__((ext_vector_type(4)))  unsigned int u32x4;
typedef __attribute__((ext_vector_type(4)))  int          v4i;

union V16 { u32x4 q[2]; v16bf v; };

#define QDIM        32
#define HDIM        512
#define NTRIL       528
#define ROWS        128                       // batch rows per block
#define ACT_STRIDE  520                       // bf16 elems/row (512 + pad)
#define STRIP_STRIDE 520                      // bf16 elems per strip row
#define STRIP_BYTES (16 * STRIP_STRIDE * 2)   // 16640 B per weight strip
#define ACT_OFF     (2 * STRIP_BYTES)         // 33280
#define LP_OFF      ACT_OFF                   // L_params reuses act region
#define LP_STRIDE   532                       // f32 elems per Lp row (528+pad)
#define BIAS_OFF    (LP_OFF + ROWS * LP_STRIDE * 4)     // 305,664
#define BIAS_FLOATS (4 * 512 + NTRIL)                   // 2576
#define SMEM_BYTES  (BIAS_OFF + BIAS_FLOATS * 4)        // 315,968 <= 320 KB

// Async global->LDS DMA path (CDNA5), guarded so either toolchain compiles.
#if defined(__gfx1250__) && \
    __has_builtin(__builtin_amdgcn_global_load_async_to_lds_b128)
#define USE_ASYNC 1
#else
#define USE_ASYNC 0
#endif

__device__ __forceinline__ void async_wait_all() {
#if defined(__gfx1250__)
#if __has_builtin(__builtin_amdgcn_s_wait_asynccnt)
  __builtin_amdgcn_s_wait_asynccnt(0);
#else
  asm volatile("s_wait_asynccnt 0" ::: "memory");
#endif
#endif
}

__device__ __forceinline__ v16bf lds_frag(const char* p0, const char* p1) {
  V16 t;
  t.q[0] = *(const u32x4*)p0;
  t.q[1] = *(const u32x4*)p1;
  return t.v;
}

__device__ __forceinline__ v8f wmma_bf16(v16bf a, v16bf b, v8f c) {
  return __builtin_amdgcn_wmma_f32_16x16x32_bf16(false, a, false, b,
                                                 (short)0, c, false, false);
}

// ---- weight strip staging (one 16-row x K bf16 N-panel) -------------------
#if USE_ASYNC
template <int KS>
__device__ __forceinline__ void strip_issue_async(const __bf16* Wt, int nt,
                                                  char* buf, int tid) {
  const int K = KS * 32;
  const int NV4 = 16 * K / 8;
#pragma unroll
  for (int it = 0; it < 4; ++it) {
    int i = tid + it * 256;
    if (i < NV4) {
      int r = i / (K / 8), c = i % (K / 8);
      void* g = (void*)(Wt + (size_t)nt * 16 * K + (size_t)r * K + c * 8);
      void* l = (void*)(buf + r * (STRIP_STRIDE * 2) + c * 16);
      __builtin_amdgcn_global_load_async_to_lds_b128(
          (__attribute__((address_space(1))) v4i*)g,
          (__attribute__((address_space(3))) v4i*)l, 0, 0);
    }
  }
}
#else
template <int KS>
__device__ __forceinline__ void strip_load_regs(const __bf16* Wt, int nt,
                                                u32x4 (&pf)[4], int tid) {
  const int K = KS * 32;
  const int NV4 = 16 * K / 8;
#pragma unroll
  for (int it = 0; it < 4; ++it) {
    int i = tid + it * 256;
    if (i < NV4) {
      int r = i / (K / 8), c = i % (K / 8);
      pf[it] = *(const u32x4*)(Wt + (size_t)nt * 16 * K + (size_t)r * K + c * 8);
    }
  }
}
template <int KS>
__device__ __forceinline__ void strip_store_regs(char* buf,
                                                 const u32x4 (&pf)[4], int tid) {
  const int K = KS * 32;
  const int NV4 = 16 * K / 8;
#pragma unroll
  for (int it = 0; it < 4; ++it) {
    int i = tid + it * 256;
    if (i < NV4) {
      int r = i / (K / 8), c = i % (K / 8);
      *(u32x4*)(buf + r * (STRIP_STRIDE * 2) + c * 16) = pf[it];
    }
  }
}
#endif

// Preload this wave's 16-row activation panel (K = KS*32) into A-fragments.
template <int KS>
__device__ __forceinline__ void preload_act(v16bf (&A)[KS], const char* smem,
                                            int w, int l) {
  const char* act = smem + ACT_OFF;
  const int row = w * 16 + (l & 15);
  const char* rbase = act + row * (ACT_STRIDE * 2);
#pragma unroll
  for (int ks = 0; ks < KS; ++ks) {
    const char* p0 = rbase + ks * 64 + ((l >> 4) * 16);
    A[ks] = lds_frag(p0, p0 + 32);
  }
}

// Input layer A-fragment straight from global q (f32 -> bf16 in-register).
__device__ __forceinline__ void load_q_frag(v16bf (&A)[1],
                                            const float* __restrict__ q,
                                            int gRow0, int w, int l) {
  const int row = gRow0 + w * 16 + (l & 15);
  const float* p = q + (size_t)row * QDIM;
  const int c0 = (l >> 4) * 8;
  V16 t;
#pragma unroll
  for (int j = 0; j < 8; ++j) t.v[j] = (__bf16)p[c0 + j];
#pragma unroll
  for (int j = 0; j < 8; ++j) t.v[8 + j] = (__bf16)p[16 + c0 + j];
  A[0] = t.v;
}

// One GEMM layer: [16 rows] x [K = KS*32] x [NT*16 cols], weights Wt[N][K]
// bf16 (K contiguous). Double-buffered LDS weight strips.
template <int KS, int NT, bool RELU, bool RES, bool F32OUT>
__device__ __forceinline__ void run_layer(const __bf16* __restrict__ Wt,
                                          const float* __restrict__ biasL,
                                          char* smem, const v16bf (&A)[KS],
                                          int w, int l, int tid) {
  char* strip0 = smem;
  char* strip1 = smem + STRIP_BYTES;
  __bf16* act = (__bf16*)(smem + ACT_OFF);
  float*  lp  = (float*)(smem + LP_OFF);

  // Stage strip 0 (N-tile 0).
#if USE_ASYNC
  strip_issue_async<KS>(Wt, 0, strip0, tid);
  async_wait_all();
#else
  {
    u32x4 pf[4];
    strip_load_regs<KS>(Wt, 0, pf, tid);
    strip_store_regs<KS>(strip0, pf, tid);
  }
#endif
  __syncthreads();

  for (int nt = 0; nt < NT; ++nt) {
    char* sn = ((nt + 1) & 1) ? strip1 : strip0;
#if USE_ASYNC
    if (nt + 1 < NT) strip_issue_async<KS>(Wt, nt + 1, sn, tid);
#else
    u32x4 pf[4];
    if (nt + 1 < NT) strip_load_regs<KS>(Wt, nt + 1, pf, tid);
#endif

    const char* sb = (nt & 1) ? strip1 : strip0;
    float bv = biasL[nt * 16 + (l & 15)];  // LDS read: stays off LOADcnt
    v8f acc;
#pragma unroll
    for (int i = 0; i < 8; ++i) acc[i] = bv;

    // B-fragment per lane: col = l&15 of this N-tile, bf16 at
    // K = (l>>4)*8 + 0..7 and 16 + (l>>4)*8 + 0..7 within the slice.
#pragma unroll
    for (int ks = 0; ks < KS; ++ks) {
      const char* p0 =
          sb + (l & 15) * (STRIP_STRIDE * 2) + ks * 64 + ((l >> 4) * 16);
      v16bf B = lds_frag(p0, p0 + 32);
      acc = wmma_bf16(A[ks], B, acc);
    }

    // Fused epilogue. D layout: VGPR i -> M = i + (l>>4)*8, N = l&15.
#pragma unroll
    for (int i = 0; i < 8; ++i) {
      int m   = (l >> 4) * 8 + i;
      int col = nt * 16 + (l & 15);
      float v = acc[i];
      if (RELU) v = v > 0.0f ? v : 0.0f;
      if (RES)  v += (float)act[(w * 16 + m) * ACT_STRIDE + col];
      if (F32OUT) lp[(w * 16 + m) * LP_STRIDE + col] = v;
      else        act[(w * 16 + m) * ACT_STRIDE + col] = (__bf16)v;
    }

#if USE_ASYNC
    if (nt + 1 < NT) async_wait_all();  // drain DMA before publishing strip
#else
    if (nt + 1 < NT) strip_store_regs<KS>(sn, pf, tid);
#endif
    __syncthreads();
  }
}

// Fragments of the dense 32x32 bf16 L (row stride 32 elems = 64 B).
__device__ __forceinline__ v16bf ld_A_dense(const char* Ld, int rowOff, int l) {
  const char* p0 = Ld + (rowOff + (l & 15)) * 64 + ((l >> 4) * 16);
  return lds_frag(p0, p0 + 32);
}
// B = L^T block: lane holds row (colOff + l&15) of L, K = (l>>4)*8 .. +7, 16 + (l>>4)*8 .. +7.
__device__ __forceinline__ v16bf ld_B_dense(const char* Ld, int colOff, int l) {
  const char* p0 = Ld + (colOff + (l & 15)) * 64 + ((l >> 4) * 16);
  return lds_frag(p0, p0 + 32);
}

__global__ __launch_bounds__(256) void mlp_chol_kernel(
    const float* __restrict__ q,
    const float* __restrict__ b_in, const float* __restrict__ b_h1,
    const float* __restrict__ b_h2, const float* __restrict__ b_out,
    const float* __restrict__ b_o,
    const __bf16* __restrict__ wsb, float* __restrict__ out) {
  extern __shared__ char smem[];
  const int tid = threadIdx.x;
  const int w = tid >> 5;
  const int l = tid & 31;
  const int gRow0 = blockIdx.x * ROWS;

  const __bf16* Wt_in  = wsb;                       // [512][32]
  const __bf16* Wt_h1  = Wt_in + 512 * 32;          // [512][512]
  const __bf16* Wt_h2  = Wt_h1 + 512 * 512;
  const __bf16* Wt_out = Wt_h2 + 512 * 512;
  const __bf16* Wt_o   = Wt_out + 512 * 512;        // [528][512]

  // Stage all biases into LDS once (keeps compute loop off LOADcnt).
  float* biasLds = (float*)(smem + BIAS_OFF);
  for (int i = tid; i < 512; i += 256) {
    biasLds[i]        = b_in[i];
    biasLds[512 + i]  = b_h1[i];
    biasLds[1024 + i] = b_h2[i];
    biasLds[1536 + i] = b_out[i];
  }
  for (int i = tid; i < NTRIL; i += 256) biasLds[2048 + i] = b_o[i];
  __syncthreads();

  {  // x = relu(q @ W_in + b_in), K = 32
    v16bf A[1];
    load_q_frag(A, q, gRow0, w, l);
    run_layer<1, 32, true, false, false>(Wt_in, biasLds, smem, A, w, l, tid);
  }
  {
    v16bf A[16];
    preload_act(A, smem, w, l);   // wave-private rows: no barrier needed
    run_layer<16, 32, true, true, false>(Wt_h1, biasLds + 512, smem, A, w, l, tid);
    preload_act(A, smem, w, l);
    run_layer<16, 32, true, true, false>(Wt_h2, biasLds + 1024, smem, A, w, l, tid);
    preload_act(A, smem, w, l);
    run_layer<16, 32, false, false, false>(Wt_out, biasLds + 1536, smem, A, w, l, tid);
    preload_act(A, smem, w, l);   // embed -> registers; act region now dead
    run_layer<16, 33, false, false, true>(Wt_o, biasLds + 2048, smem, A, w, l, tid);
  }

  // ---- M = L @ L^T, one sample at a time per wave, 4 bf16 WMMAs each ----
  __syncthreads();  // strips dead; reuse low LDS as wave-private L scratch
  char* Ld = smem + w * 2048;  // 32x32 bf16 dense L
  const float* lpbase = (const float*)(smem + LP_OFF);

  for (int s = 0; s < 16; ++s) {
    const int rowL = w * 16 + s;
    const float* lprow = lpbase + rowL * LP_STRIDE;
#pragma unroll 4
    for (int r = 0; r < 32; ++r) {
      int c = l;
      float v;
      if (c > r)       v = 0.0f;
      else if (c == r) v = lprow[r] + 2.0f;                     // diag + BIAS
      else             v = lprow[QDIM + (r * (r - 1)) / 2 + c]; // strict tril
      ((__bf16*)Ld)[r * 32 + c] = (__bf16)v;
    }
    // LDS ops from the same wave are in-order; fragments read fresh data.
    v16bf A0 = ld_A_dense(Ld, 0, l);
    v16bf A1 = ld_A_dense(Ld, 16, l);
    v16bf B0 = ld_B_dense(Ld, 0, l);
    v16bf B1 = ld_B_dense(Ld, 16, l);
    v8f z;
#pragma unroll
    for (int i = 0; i < 8; ++i) z[i] = 0.0f;
    v8f t00 = wmma_bf16(A0, B0, z);
    v8f t01 = wmma_bf16(A0, B1, z);
    v8f t10 = wmma_bf16(A1, B0, z);
    v8f t11 = wmma_bf16(A1, B1, z);

    float* o = out + ((size_t)gRow0 + rowL) * (QDIM * QDIM);
#pragma unroll
    for (int i = 0; i < 8; ++i) {
      int m = (l >> 4) * 8 + i;
      int n = l & 15;
      o[m * 32 + n]             = t00[i];
      o[m * 32 + 16 + n]        = t01[i];
      o[(16 + m) * 32 + n]      = t10[i];
      o[(16 + m) * 32 + 16 + n] = t11[i];
    }
  }
}

// Transpose + f32->bf16: out[n*K + k] = (bf16) W[k*N + n]
__global__ void cvt_transpose_kernel(const float* __restrict__ W,
                                     __bf16* __restrict__ out, int K, int N) {
  int i = blockIdx.x * 256 + threadIdx.x;
  if (i < N * K) {
    int n = i / K, k = i % K;
    out[i] = (__bf16)W[(size_t)k * N + n];
  }
}

extern "C" void kernel_launch(void* const* d_in, const int* in_sizes, int n_in,
                              void* d_out, int out_size, void* d_ws,
                              size_t ws_size, hipStream_t stream) {
  const float* q     = (const float*)d_in[0];
  const float* W_in  = (const float*)d_in[1];
  const float* b_in  = (const float*)d_in[2];
  const float* W_h1  = (const float*)d_in[3];
  const float* b_h1  = (const float*)d_in[4];
  const float* W_h2  = (const float*)d_in[5];
  const float* b_h2  = (const float*)d_in[6];
  const float* W_out = (const float*)d_in[7];
  const float* b_out = (const float*)d_in[8];
  const float* W_o   = (const float*)d_in[9];
  const float* b_o   = (const float*)d_in[10];
  float* out = (float*)d_out;

  __bf16* wsb = (__bf16*)d_ws;
  __bf16* Wt_in  = wsb;
  __bf16* Wt_h1  = Wt_in + 512 * 32;
  __bf16* Wt_h2  = Wt_h1 + 512 * 512;
  __bf16* Wt_out = Wt_h2 + 512 * 512;
  __bf16* Wt_o   = Wt_out + 512 * 512;

  cvt_transpose_kernel<<<(512 * 32 + 255) / 256, 256, 0, stream>>>(W_in, Wt_in, 32, 512);
  cvt_transpose_kernel<<<(512 * 512 + 255) / 256, 256, 0, stream>>>(W_h1, Wt_h1, 512, 512);
  cvt_transpose_kernel<<<(512 * 512 + 255) / 256, 256, 0, stream>>>(W_h2, Wt_h2, 512, 512);
  cvt_transpose_kernel<<<(512 * 512 + 255) / 256, 256, 0, stream>>>(W_out, Wt_out, 512, 512);
  cvt_transpose_kernel<<<(512 * NTRIL + 255) / 256, 256, 0, stream>>>(W_o, Wt_o, 512, NTRIL);

  const int nBlocks = 65536 / ROWS;  // 512
  mlp_chol_kernel<<<nBlocks, 256, SMEM_BYTES, stream>>>(
      q, b_in, b_h1, b_h2, b_out, b_o, (const __bf16*)wsb, out);
}
